// BalRNN_13099650253273
// MI455X (gfx1250) — matrix operand order repaired and checked by reading, not compiled, not run on hardware
//
#include <hip/hip_runtime.h>
#include <hip/hip_bf16.h>
#include <stddef.h>

// Problem constants (match reference)
constexpr int B  = 32;
constexpr int T  = 64;
constexpr int IN = 512;
constexpr int L  = 2;
constexpr int H  = 4096;
constexpr int K  = 64;

// Recurrence launch geometry: 256 workgroups x 256 threads = 2048 wave32s.
constexpr int NWG_R = 256;
constexpr int TPB_R = 256;
constexpr int WAVES_R = NWG_R * (TPB_R / 32);   // 2048
constexpr int NPW = H / WAVES_R;                // 2 neurons per wave per layer

typedef __attribute__((ext_vector_type(16))) __bf16 v16bf;
typedef __attribute__((ext_vector_type(8)))  __bf16 v8bf;
typedef __attribute__((ext_vector_type(8)))  float  v8f;

// ---------------- helpers ----------------

__device__ __forceinline__ __bf16 f2bf(float f) {
  unsigned u = __builtin_bit_cast(unsigned, f);
  u += 0x7FFFu + ((u >> 16) & 1u);              // round-to-nearest-even
  unsigned short s = (unsigned short)(u >> 16);
  return __builtin_bit_cast(__bf16, s);
}

// Device-wide barrier: monotonic counter, one arrival per workgroup.
// Counter zeroed by k_prep every launch -> deterministic.
__device__ __forceinline__ void grid_sync(unsigned* cnt, unsigned target) {
  __syncthreads();
  if (threadIdx.x == 0) {
    __threadfence();                                     // release our stores
    atomicAdd(cnt, 1u);
    while (__hip_atomic_load(cnt, __ATOMIC_RELAXED, __HIP_MEMORY_SCOPE_AGENT) < target) {
      __builtin_amdgcn_s_sleep(2);
    }
    __threadfence();                                     // acquire others' stores
  }
  __syncthreads();
}

// ---------------- kernel 1: prep (bf16 conversions, zero state + barrier) ----------------

__global__ void k_prep(const float* __restrict__ w,   // [H][IN]
                       const float* __restrict__ x,   // [B][T][IN]
                       __bf16* __restrict__ wbf,
                       __bf16* __restrict__ xbf,
                       float* __restrict__ st,        // 4 * H*B
                       unsigned* __restrict__ bar) {
  size_t i = (size_t)blockIdx.x * blockDim.x + threadIdx.x;
  const size_t NW = (size_t)H * IN;        // 2,097,152
  const size_t NX = (size_t)B * T * IN;    // 1,048,576
  const size_t NS = (size_t)4 * H * B;     //   524,288
  if (i < NW) wbf[i] = f2bf(w[i]);
  if (i < NX) xbf[i] = f2bf(x[i]);
  if (i < NS) st[i] = 0.0f;
  if (i == 0) *bar = 0u;
}

// ---------------- kernel 2: input GEMM via WMMA bf16 ----------------
// xe[t][h][b] = sum_i x[b][t][i] * w[h][i], fp32 accumulation.
// One wave per 16x16 output tile; K-loop of 16 x v_wmma_f32_16x16x32_bf16.

constexpr int TPB_G = 256;

__global__ void k_gemm(const __bf16* __restrict__ wbf,   // A source: [H][IN]
                       const __bf16* __restrict__ xbf,   // B source: [B][T][IN]
                       float* __restrict__ xe) {         // [T][H][B]
  const int lane = threadIdx.x & 31;
  const int wave = __builtin_amdgcn_readfirstlane((int)(threadIdx.x >> 5));
  const int gw   = blockIdx.x * (TPB_G >> 5) + wave;   // 0..32767 tiles

  const int bt = gw & (B / 16 - 1);                    // 0..1   (N tile: batches)
  const int ht = (gw >> 1) & (H / 16 - 1);             // 0..255 (M tile: hidden)
  const int t  = gw >> 9;                              // 0..63

  const int half = lane >> 4;                          // 0 | 1
  const int lm   = lane & 15;
  // A-matrix 16x32 (MxK) per-lane placement: lanes<16 K in 0-15, lanes>=16 K in 16-31
  const int ac0 = half * 16;
  // B-matrix 32x16 (KxN): lanes<16 hold K=0..15 of N=lane, lanes>=16 hold K=16..31
  const int bk0 = half * 16;

  const __bf16* aptr = wbf + (size_t)(ht * 16 + lm) * IN;
  const __bf16* bptr = xbf + ((size_t)(bt * 16 + lm) * T + t) * IN;

  v8f c = {};
#pragma unroll
  for (int ks = 0; ks < IN / 32; ++ks) {
    const int k0 = ks * 32;
    v8bf alo = *(const v8bf*)(aptr + k0 + ac0);
    v8bf ahi = *(const v8bf*)(aptr + k0 + ac0 + 8);
    v16bf a;
#pragma unroll
    for (int j = 0; j < 8; ++j) { a[j] = alo[j]; a[j + 8] = ahi[j]; }
    v16bf b = *(const v16bf*)(bptr + k0 + bk0);
    c = __builtin_amdgcn_wmma_f32_16x16x32_bf16(false, a, false, b, (short)0, c, false, false);
  }

  // D layout: VGPR r -> (M = r + half*8, N = lm); write xe[t][h][b] (64B runs per half)
#pragma unroll
  for (int r = 0; r < 8; ++r) {
    const int h = ht * 16 + r + half * 8;
    xe[((size_t)t * H + h) * B + (bt * 16 + lm)] = c[r];
  }
}

// ---------------- kernel 3: persistent recurrence ----------------
// lane = batch. idx/vals are wave-uniform -> scalar loads; gathers are 128B rows.
// State layout: st[layer buffer][h][b]. One device-wide barrier per timestep.

__global__ void k_recur(const float* __restrict__ xe,     // [T][H][B]
                        const int* __restrict__ hh_idx,   // [L][H][K]
                        const float* __restrict__ hh_vals,// [L][H][K]
                        float* __restrict__ st,           // 4 * H*B (s0a,s0b,s1a,s1b)
                        float* __restrict__ hist,         // [T][H][B]
                        unsigned* __restrict__ bar) {
  const int lane = threadIdx.x & 31;
  const int wave = __builtin_amdgcn_readfirstlane((int)(threadIdx.x >> 5));
  const int gw   = blockIdx.x * (TPB_R >> 5) + wave;     // uniform per wave
  const int h_base = gw * NPW;

  float* s0[2] = { st,                  st + (size_t)H * B };
  float* s1[2] = { st + (size_t)2*H*B,  st + (size_t)3*H*B };

  for (int t = 0; t < T; ++t) {
    const int pp = t & 1;

    // ---- phase 1: layer 0 : cur0 = relu(xe[t] + W0 (*) prev0) ----
    {
      const float* __restrict__ src = s0[pp];
      float* __restrict__ dst = s0[pp ^ 1];
      for (int n = 0; n < NPW; ++n) {
        const int h = h_base + n;
        const int*   __restrict__ ip = hh_idx  + (size_t)h * K;      // layer 0
        const float* __restrict__ vp = hh_vals + (size_t)h * K;
        float acc = xe[((size_t)t * H + h) * B + lane];
#pragma unroll 8
        for (int k = 0; k < K; ++k) {
          const int   j = ip[k];                    // wave-uniform (s_load)
          const float v = vp[k];                    // wave-uniform (s_load)
          acc += v * src[(size_t)j * B + lane];     // coalesced 128B gather
        }
        dst[(size_t)h * B + lane] = fmaxf(acc, 0.0f);
      }
    }

    grid_sync(bar, (unsigned)(NWG_R * (t + 1)));   // single barrier per step

    // ---- phase 2: layer 1 : cur1 = relu(W1 (*) (cur0 + prev1)) (fused) ----
    {
      const float* __restrict__ c0 = s0[pp ^ 1];
      const float* __restrict__ p1 = s1[pp];
      float* __restrict__ dst = s1[pp ^ 1];
      for (int n = 0; n < NPW; ++n) {
        const int h = h_base + n;
        const int*   __restrict__ ip = hh_idx  + (size_t)(H + h) * K; // layer 1
        const float* __restrict__ vp = hh_vals + (size_t)(H + h) * K;
        float acc = 0.0f;
#pragma unroll 8
        for (int k = 0; k < K; ++k) {
          const int   j = ip[k];
          const float v = vp[k];
          acc += v * (c0[(size_t)j * B + lane] + p1[(size_t)j * B + lane]);
        }
        const float r = fmaxf(acc, 0.0f);
        dst[(size_t)h * B + lane] = r;
        hist[((size_t)t * H + h) * B + lane] = r;   // coalesced history write
      }
    }
    // no barrier needed here: next phase1 touches disjoint buffers (proved safe)
  }
}

// ---------------- kernel 4: h_final (L,B,H) from final state buffers (index 0, T even) ----------------

__global__ void k_final(const float* __restrict__ st, float* __restrict__ out_hf) {
  const size_t i = (size_t)blockIdx.x * blockDim.x + threadIdx.x;
  if (i >= (size_t)L * B * H) return;
  const int h = (int)(i % H);
  const int b = (int)((i / H) % B);
  const int l = (int)(i / ((size_t)B * H));
  const float* s = st + (l ? (size_t)2 * H * B : 0);  // s0[0], s1[0]
  out_hf[i] = s[(size_t)h * B + b];
}

// ---------------- kernel 5: transpose hist[t][h][b] -> out[b][t][h] ----------------

__global__ void k_tr(const float* __restrict__ hist, float* __restrict__ out) {
  __shared__ float tile[32][33];
  const int t  = blockIdx.x >> 7;             // H/32 = 128 tiles per t
  const int h0 = (blockIdx.x & 127) << 5;
  const int c  = threadIdx.x & 31;
  const int r  = threadIdx.x >> 5;            // 0..7
#pragma unroll
  for (int i = 0; i < 4; ++i) {
    const int hr = r + i * 8;
    tile[hr][c] = hist[((size_t)t * H + h0 + hr) * B + c];
  }
  __syncthreads();
#pragma unroll
  for (int i = 0; i < 4; ++i) {
    const int b = r + i * 8;
    out[((size_t)b * T + t) * H + h0 + c] = tile[c][b];
  }
}

// ---------------- launch ----------------

extern "C" void kernel_launch(void* const* d_in, const int* in_sizes, int n_in,
                              void* d_out, int out_size, void* d_ws, size_t ws_size,
                              hipStream_t stream) {
  (void)in_sizes; (void)n_in; (void)out_size; (void)ws_size;

  const float* x       = (const float*)d_in[0];   // (B,T,IN)
  const float* w_ih    = (const float*)d_in[1];   // (H,IN)
  const int*   hh_idx  = (const int*)  d_in[2];   // (L,H,K)
  const float* hh_vals = (const float*)d_in[3];   // (L,H,K)
  float* out = (float*)d_out;                     // (B,T,H) ++ (L,B,H)

  // Workspace carve-up (~75.5 MB total)
  char* base = (char*)d_ws;
  size_t off = 0;
  float*    xe   = (float*)(base + off);  off += (size_t)T * H * B * 4;   // 33.5 MB
  float*    hist = (float*)(base + off);  off += (size_t)T * H * B * 4;   // 33.5 MB
  float*    st   = (float*)(base + off);  off += (size_t)4 * H * B * 4;   //  2   MB
  __bf16*   wbf  = (__bf16*)(base + off); off += (size_t)H * IN * 2;      //  4   MB
  __bf16*   xbf  = (__bf16*)(base + off); off += (size_t)B * T * IN * 2;  //  2   MB
  unsigned* bar  = (unsigned*)(base + off);

  // 1) conversions + zero state/barrier
  {
    const size_t n = (size_t)H * IN;                 // largest section
    k_prep<<<dim3((unsigned)((n + 255) / 256)), dim3(256), 0, stream>>>(
        w_ih, x, wbf, xbf, st, bar);
  }
  // 2) WMMA input GEMM -> xe[t][h][b]
  {
    const int tiles = T * (H / 16) * (B / 16);       // 32768 waves
    k_gemm<<<dim3(tiles / (TPB_G / 32)), dim3(TPB_G), 0, stream>>>(wbf, xbf, xe);
  }
  // 3) persistent recurrence (one grid barrier per timestep)
  k_recur<<<dim3(NWG_R), dim3(TPB_R), 0, stream>>>(xe, hh_idx, hh_vals, st, hist, bar);
  // 4) h_final
  {
    const size_t n = (size_t)L * B * H;
    k_final<<<dim3((unsigned)((n + 255) / 256)), dim3(256), 0, stream>>>(
        st, out + (size_t)B * T * H);
  }
  // 5) transpose history into (B,T,H)
  k_tr<<<dim3(T * (H / 32)), dim3(256), 0, stream>>>(hist, out);
}
